// DGCNN_61959198212099
// MI455X (gfx1250) — compile-verified, matrix operand order checked by reading, not run on hardware
//
#include <hip/hip_runtime.h>

// DGCNN forward for MI455X (gfx1250, wave32).
// Edge-conv GEMMs on v_wmma_f32_16x16x32_f16 with branch-free vectorized
// fragment loads: activations mirrored in f16, weights pre-swizzled into the
// native B-fragment layout. BN via two-pass (sum/sumsq atomics, then fused
// scale/shift + ReLU + max-over-k with monotone uint atomicMax).

typedef __attribute__((ext_vector_type(16))) _Float16 v16h;
typedef __attribute__((ext_vector_type(8)))  _Float16 v8h;
typedef __attribute__((ext_vector_type(8)))  float    v8f;

#define BATCH 16
#define NPTS  1024
#define KNN   20
#define MROWS (BATCH * NPTS * KNN)   // 327680, divisible by 512
#define BN_EPS 1e-5f

static_assert(MROWS % 512 == 0, "row tiling");

__device__ __forceinline__ float dist4(float4 p, float4 q, float s)
{
    s = fmaf(p.x - q.x, p.x - q.x, s);
    s = fmaf(p.y - q.y, p.y - q.y, s);
    s = fmaf(p.z - q.z, p.z - q.z, s);
    s = fmaf(p.w - q.w, p.w - q.w, s);
    return s;
}

// ---------------------------------------------------------------------------
// kNN: one wave32 per (b,n); 32 candidate distances per lane held in VGPRs
// (fully unrolled), 20 rounds of wave-wide argmin with key=(dist_bits<<32)|j
// so ties resolve to the lowest index like jax.lax.top_k.
// ---------------------------------------------------------------------------
template<int CIN>
__global__ __launch_bounds__(256) void dgcnn_knn_kernel(
    const float* __restrict__ x, int* __restrict__ idx)
{
    const int lane = threadIdx.x & 31;
    const int wave = threadIdx.x >> 5;
    const int row  = blockIdx.x * 8 + wave;          // b*N + n
    const int b    = row >> 10;                      // row / NPTS
    const int n    = row & (NPTS - 1);
    const float* xi = x + (size_t)row * CIN;
    const float* xb = x + (size_t)b * NPTS * CIN;

    float d[32];
    #pragma unroll
    for (int t = 0; t < 32; ++t) d[t] = 0.f;

    if constexpr (CIN % 16 == 0) {
        for (int c0 = 0; c0 < CIN; c0 += 16) {       // query chunk in registers
            const float4 p0 = *(const float4*)(xi + c0);
            const float4 p1 = *(const float4*)(xi + c0 + 4);
            const float4 p2 = *(const float4*)(xi + c0 + 8);
            const float4 p3 = *(const float4*)(xi + c0 + 12);
            #pragma unroll
            for (int t = 0; t < 32; ++t) {
                const float* xj = xb + (t * 32 + lane) * CIN + c0;
                float s = d[t];
                s = dist4(p0, *(const float4*)(xj),      s);
                s = dist4(p1, *(const float4*)(xj + 4),  s);
                s = dist4(p2, *(const float4*)(xj + 8),  s);
                s = dist4(p3, *(const float4*)(xj + 12), s);
                d[t] = s;
            }
        }
    } else {                                          // CIN == 3
        const float xi0 = xi[0], xi1 = xi[1], xi2 = xi[2];
        #pragma unroll
        for (int t = 0; t < 32; ++t) {
            const float* xj = xb + (t * 32 + lane) * CIN;
            const float d0 = xi0 - xj[0], d1 = xi1 - xj[1], d2 = xi2 - xj[2];
            d[t] = fmaf(d0, d0, fmaf(d1, d1, d2 * d2));
        }
    }

    if ((n & 31) == lane) {                           // exclude self
        const int ts = n >> 5;
        #pragma unroll
        for (int t = 0; t < 32; ++t)
            if (t == ts) d[t] = __builtin_inff();
    }

    for (int kk = 0; kk < KNN; ++kk) {
        float best = d[0]; int bt = 0;
        #pragma unroll
        for (int t = 1; t < 32; ++t)
            if (d[t] < best) { best = d[t]; bt = t; }
        unsigned long long key =
            ((unsigned long long)__float_as_uint(best) << 32) |
            (unsigned)(bt * 32 + lane);
        #pragma unroll
        for (int off = 16; off > 0; off >>= 1) {
            unsigned long long o = __shfl_xor(key, off);
            if (o < key) key = o;
        }
        const int jstar = (int)(unsigned)(key & 0xffffffffu);
        if ((jstar & 31) == lane) {
            const int ts = jstar >> 5;
            #pragma unroll
            for (int t = 0; t < 32; ++t)
                if (t == ts) d[t] = __builtin_inff();
        }
        if (lane == 0) idx[row * KNN + kk] = jstar;
    }
}

// ---------------------------------------------------------------------------
// Weight swizzle: W (Ktot x Nout, f32) -> f16 in native B-fragment order:
// out[((ct*KCHT + ck)*32 + lane)*16 + e], zero-padded in K and N.
// ---------------------------------------------------------------------------
__global__ void dgcnn_wswizzle_kernel(const float* __restrict__ W, int Ktot,
                                      int Nout, int KCHT,
                                      _Float16* __restrict__ out, int total)
{
    const int tid = blockIdx.x * blockDim.x + threadIdx.x;
    if (tid >= total) return;
    const int e    = tid & 15;
    const int lane = (tid >> 4) & 31;
    const int rest = tid >> 9;
    const int ck   = rest % KCHT;
    const int ct   = rest / KCHT;
    const int c    = ck * 32 + ((lane >> 4) << 3) + ((e >> 3) << 4) + (e & 7);
    const int n    = ct * 16 + (lane & 15);
    const float v  = (c < Ktot && n < Nout) ? W[c * Nout + n] : 0.f;
    out[tid] = (_Float16)v;
}

__global__ void dgcnn_f32tof16_kernel(const float* __restrict__ in,
                                      _Float16* __restrict__ out, int n)
{
    const int i = blockIdx.x * blockDim.x + threadIdx.x;
    if (i < n) out[i] = (_Float16)in[i];
}

__global__ void dgcnn_fill_kernel(float* __restrict__ p, float v, size_t n)
{
    const size_t i = (size_t)blockIdx.x * blockDim.x + threadIdx.x;
    if (i < n) p[i] = v;
}

// ---------------------------------------------------------------------------
// One 16x16 f32 tile of h = [x_i - x_j | x_j] @ W.  For CIN % 32 == 0 the
// concat boundary is chunk-aligned -> two branch-free K loops; per-lane data
// is two contiguous 8-half groups (16 B vector loads + v_pk_sub_f16).
// ---------------------------------------------------------------------------
template<int CIN>
__device__ __forceinline__ v8f dgcnn_h_tile(const _Float16* __restrict__ xh,
                                            const int* __restrict__ idx,
                                            const _Float16* __restrict__ Wsw,
                                            int rowbase)
{
    const int lane  = threadIdx.x & 31;
    const int m     = lane & 15;
    const int kbase = (lane >> 4) << 3;
    const int r     = rowbase + m;                   // enumerates (b, n, kk)
    const int bi    = r / (NPTS * KNN);
    const int rem   = r - bi * (NPTS * KNN);
    const int n     = rem / KNN;
    const int j     = idx[r];
    const _Float16* xi = xh + (bi * NPTS + n) * CIN;
    const _Float16* xj = xh + (bi * NPTS + j) * CIN;

    v8f acc = {};
    if constexpr (CIN % 32 == 0) {
        constexpr int KCH = CIN / 32;
        #pragma unroll
        for (int ck = 0; ck < KCH; ++ck) {           // x_i - x_j half
            const int c0 = ck * 32 + kbase;
            const v8h d0 = *(const v8h*)(xi + c0)      - *(const v8h*)(xj + c0);
            const v8h d1 = *(const v8h*)(xi + c0 + 16) - *(const v8h*)(xj + c0 + 16);
            const v16h a = __builtin_shufflevector(d0, d1,
                0,1,2,3,4,5,6,7,8,9,10,11,12,13,14,15);
            const v16h bf = *(const v16h*)(Wsw + ((ck * 32 + lane) << 4));
            acc = __builtin_amdgcn_wmma_f32_16x16x32_f16(
                      false, a, false, bf, (short)0, acc, false, false);
        }
        #pragma unroll
        for (int ck = 0; ck < KCH; ++ck) {           // x_j half
            const int c0 = ck * 32 + kbase;
            const v8h a0 = *(const v8h*)(xj + c0);
            const v8h a1 = *(const v8h*)(xj + c0 + 16);
            const v16h a = __builtin_shufflevector(a0, a1,
                0,1,2,3,4,5,6,7,8,9,10,11,12,13,14,15);
            const v16h bf = *(const v16h*)(Wsw + (((KCH + ck) * 32 + lane) << 4));
            acc = __builtin_amdgcn_wmma_f32_16x16x32_f16(
                      false, a, false, bf, (short)0, acc, false, false);
        }
    } else {                                          // CIN == 3, K = 6 (padded)
        v16h a = {};
        if (kbase == 0) {                             // lanes 16..31 carry zeros
            const _Float16 a0 = xi[0], a1 = xi[1], a2 = xi[2];
            const _Float16 b0 = xj[0], b1 = xj[1], b2 = xj[2];
            a[0] = a0 - b0; a[1] = a1 - b1; a[2] = a2 - b2;
            a[3] = b0;      a[4] = b1;      a[5] = b2;
        }
        const v16h bf = *(const v16h*)(Wsw + (lane << 4));
        acc = __builtin_amdgcn_wmma_f32_16x16x32_f16(
                  false, a, false, bf, (short)0, acc, false, false);
    }
    return acc;
}

// Pass 1: per-channel sum / sumsq of h (linear bias cancels in BN -> omitted).
template<int CIN>
__global__ __launch_bounds__(128) void dgcnn_stats_kernel(
    const _Float16* __restrict__ xh, const int* __restrict__ idx,
    const _Float16* __restrict__ Wsw,
    float* __restrict__ sum, float* __restrict__ sumsq)
{
    constexpr int KCHT = (CIN == 3) ? 1 : (2 * CIN) / 32;
    const int lane = threadIdx.x & 31;
    const int wave = threadIdx.x >> 5;
    const int col0 = blockIdx.y * 16;
    const _Float16* Wt = Wsw + blockIdx.y * (KCHT * 512);
    const int rowbase0 = (blockIdx.x * 4 + wave) * 128;

    float s = 0.f, sq = 0.f;
    for (int t = 0; t < 8; ++t) {
        v8f acc = dgcnn_h_tile<CIN>(xh, idx, Wt, rowbase0 + t * 16);
        #pragma unroll
        for (int v = 0; v < 8; ++v) { const float h = acc[v]; s += h; sq = fmaf(h, h, sq); }
    }
    s  += __shfl_xor(s, 16);        // lanes l and l^16 share a column
    sq += __shfl_xor(sq, 16);
    if (lane < 16) {
        atomicAdd(&sum  [col0 + lane], s);
        atomicAdd(&sumsq[col0 + lane], sq);
    }
}

__global__ void dgcnn_bn_finalize_kernel(
    const float* __restrict__ sum, const float* __restrict__ sumsq,
    const float* __restrict__ gamma, const float* __restrict__ beta,
    int Cout, float* __restrict__ scale, float* __restrict__ shift)
{
    const int c = blockIdx.x * blockDim.x + threadIdx.x;
    if (c >= Cout) return;
    const float invM = 1.0f / (float)MROWS;
    const float mean = sum[c] * invM;
    const float var  = fmaf(-mean, mean, sumsq[c] * invM);
    const float sc   = rsqrtf(var + BN_EPS) * gamma[c];
    scale[c] = sc;
    shift[c] = fmaf(-mean, sc, beta[c]);
}

// Pass 2: recompute tiles, BN+ReLU, max over k via uint atomicMax
// (post-ReLU values >= 0 -> IEEE bits order-preserving; y pre-zeroed).
template<int CIN>
__global__ __launch_bounds__(128) void dgcnn_apply_kernel(
    const _Float16* __restrict__ xh, const int* __restrict__ idx,
    const _Float16* __restrict__ Wsw, int Cout,
    const float* __restrict__ scale, const float* __restrict__ shift,
    float* __restrict__ y)
{
    constexpr int KCHT = (CIN == 3) ? 1 : (2 * CIN) / 32;
    const int lane = threadIdx.x & 31;
    const int wave = threadIdx.x >> 5;
    const int col0 = blockIdx.y * 16;
    const _Float16* Wt = Wsw + blockIdx.y * (KCHT * 512);
    const int ncol = col0 + (lane & 15);
    const int half = lane >> 4;
    const float sc = scale[ncol];
    const float sh = shift[ncol];
    const int rowbase0 = (blockIdx.x * 4 + wave) * 128;

    unsigned int* yb = (unsigned int*)y;
    int   cur_pt  = -1;            // running (b*N+n) this lane reduces over
    float cur_max = 0.f;
    for (int t = 0; t < 8; ++t) {
        const int rowbase = rowbase0 + t * 16;
        v8f acc = dgcnn_h_tile<CIN>(xh, idx, Wt, rowbase);
        #pragma unroll
        for (int v = 0; v < 8; ++v) {
            const int r  = rowbase + half * 8 + v;
            const float h = fmaxf(fmaf(acc[v], sc, sh), 0.f);
            const int pt = r / KNN;
            if (pt != cur_pt) {
                if (cur_pt >= 0 && cur_max > 0.f)
                    atomicMax(&yb[(size_t)cur_pt * Cout + ncol], __float_as_uint(cur_max));
                cur_pt = pt; cur_max = h;
            } else {
                cur_max = fmaxf(cur_max, h);
            }
        }
    }
    if (cur_pt >= 0 && cur_max > 0.f)
        atomicMax(&yb[(size_t)cur_pt * Cout + ncol], __float_as_uint(cur_max));
}

// Global max over N -> f16 (feeds FC A-fragments); coalesced across channels.
__global__ void dgcnn_pool_kernel(const float* __restrict__ y, int Cout,
                                  _Float16* __restrict__ pooledh)
{
    const int t = blockIdx.x * blockDim.x + threadIdx.x;   // b*Cout + c
    const int b = t / Cout, c = t % Cout;
    const float* p = y + (size_t)b * NPTS * Cout + c;
    float m = -__builtin_inff();
    for (int n = 0; n < NPTS; ++n) m = fmaxf(m, p[(size_t)n * Cout]);
    pooledh[t] = (_Float16)m;
}

// FC on WMMA: M=16 (batch), one wave per 16-col tile, A f16 / W pre-swizzled.
__global__ __launch_bounds__(128) void dgcnn_fc_kernel(
    const _Float16* __restrict__ A, const _Float16* __restrict__ Wsw,
    const float* __restrict__ bias, int Kdim, int Nout,
    float* __restrict__ out32, _Float16* __restrict__ out16)
{
    const int lane = threadIdx.x & 31;
    const int wave = threadIdx.x >> 5;
    const int ct   = blockIdx.x * 4 + wave;
    const int col0 = ct * 16;
    if (col0 >= Nout) return;                        // wave-uniform
    const int m     = lane & 15;
    const int kbase = (lane >> 4) << 3;
    const _Float16* Wt = Wsw + ct * ((Kdim / 32) * 512);

    v8f acc = {};
    #pragma unroll 4
    for (int ck = 0; ck < Kdim / 32; ++ck) {
        const int c0 = ck * 32 + kbase;
        const v8h a0 = *(const v8h*)(A + m * Kdim + c0);
        const v8h a1 = *(const v8h*)(A + m * Kdim + c0 + 16);
        const v16h a = __builtin_shufflevector(a0, a1,
            0,1,2,3,4,5,6,7,8,9,10,11,12,13,14,15);
        const v16h bf = *(const v16h*)(Wt + ((ck * 32 + lane) << 4));
        acc = __builtin_amdgcn_wmma_f32_16x16x32_f16(
                  false, a, false, bf, (short)0, acc, false, false);
    }
    const int half = lane >> 4;
    const int ncol = col0 + m;
    if (ncol < Nout) {
        const float bz = bias[ncol];
        #pragma unroll
        for (int v = 0; v < 8; ++v) {
            const float o = acc[v] + bz;
            const int row = half * 8 + v;
            if (out16) out16[row * Nout + ncol] = (_Float16)fmaxf(o, 0.f);
            else       out32[row * Nout + ncol] = o;
        }
    }
}

// ---------------------------------------------------------------------------
template<int CIN>
static void dgcnn_run_layer(const float* xin, const float* W, const float* g,
                            const float* be, float* xout, int Cout,
                            _Float16* xh, _Float16* Wsw, int* idxb,
                            float* stats, float* bn, hipStream_t stream)
{
    constexpr int KCHT = (CIN == 3) ? 1 : (2 * CIN) / 32;
    dgcnn_knn_kernel<CIN><<<BATCH * NPTS / 8, 256, 0, stream>>>(xin, idxb);

    const int xcnt = BATCH * NPTS * CIN;
    dgcnn_f32tof16_kernel<<<(xcnt + 255) / 256, 256, 0, stream>>>(xin, xh, xcnt);

    const int swcnt = (Cout / 16) * KCHT * 512;
    dgcnn_wswizzle_kernel<<<(swcnt + 255) / 256, 256, 0, stream>>>(
        W, 2 * CIN, Cout, KCHT, Wsw, swcnt);

    dgcnn_fill_kernel<<<8, 256, 0, stream>>>(stats, 0.f, 2048);
    dim3 gg(MROWS / 512, Cout / 16);
    dgcnn_stats_kernel<CIN><<<gg, 128, 0, stream>>>(xh, idxb, Wsw,
                                                    stats, stats + 1024);
    dgcnn_bn_finalize_kernel<<<(Cout + 255) / 256, 256, 0, stream>>>(
        stats, stats + 1024, g, be, Cout, bn, bn + 1024);

    const size_t ysz = (size_t)BATCH * NPTS * Cout;
    dgcnn_fill_kernel<<<(unsigned)((ysz + 255) / 256), 256, 0, stream>>>(xout, 0.f, ysz);
    dgcnn_apply_kernel<CIN><<<gg, 128, 0, stream>>>(xh, idxb, Wsw, Cout,
                                                    bn, bn + 1024, xout);
}

extern "C" void kernel_launch(void* const* d_in, const int* in_sizes, int n_in,
                              void* d_out, int out_size, void* d_ws, size_t ws_size,
                              hipStream_t stream)
{
    const float* pos = (const float*)d_in[0];
    const float* Wl [4] = { (const float*)d_in[1],  (const float*)d_in[5],
                            (const float*)d_in[9],  (const float*)d_in[13] };
    const float* gl [4] = { (const float*)d_in[3],  (const float*)d_in[7],
                            (const float*)d_in[11], (const float*)d_in[15] };
    const float* bel[4] = { (const float*)d_in[4],  (const float*)d_in[8],
                            (const float*)d_in[12], (const float*)d_in[16] };
    const float* Wc1 = (const float*)d_in[17];
    const float* bc1 = (const float*)d_in[18];
    const float* Wc2 = (const float*)d_in[19];
    const float* bc2 = (const float*)d_in[20];

    // Workspace carve-up (~79 MB total).
    char* ws = (char*)d_ws;
    size_t off = 0;
    auto carve = [&](size_t bytes) -> void* {
        off = (off + 255) & ~(size_t)255;
        void* p = ws + off;
        off += bytes;
        return p;
    };
    float*    bufA    = (float*)   carve((size_t)BATCH * NPTS * 1024 * 4); // 64 MB
    float*    bufB    = (float*)   carve((size_t)BATCH * NPTS * 128  * 4); //  8 MB
    _Float16* xh      = (_Float16*)carve((size_t)BATCH * NPTS * 128  * 2); //  4 MB
    _Float16* Wsw     = (_Float16*)carve((size_t)1024 * 512 * 2);          //  1 MB
    int*      idxb    = (int*)     carve((size_t)MROWS * 4);
    float*    stats   = (float*)   carve(2048 * 4);    // sum | sumsq
    float*    bn      = (float*)   carve(2048 * 4);    // scale | shift
    _Float16* pooledh = (_Float16*)carve((size_t)BATCH * 1024 * 2);
    _Float16* f1h     = (_Float16*)carve((size_t)BATCH * 512  * 2);
    if (off > ws_size) return;   // workspace too small; bail rather than corrupt

    dgcnn_run_layer<3>  (pos,  Wl[0], gl[0], bel[0], bufB, 64,
                         xh, Wsw, idxb, stats, bn, stream);
    dgcnn_run_layer<64> (bufB, Wl[1], gl[1], bel[1], bufA, 64,
                         xh, Wsw, idxb, stats, bn, stream);
    dgcnn_run_layer<64> (bufA, Wl[2], gl[2], bel[2], bufB, 128,
                         xh, Wsw, idxb, stats, bn, stream);
    dgcnn_run_layer<128>(bufB, Wl[3], gl[3], bel[3], bufA, 1024,
                         xh, Wsw, idxb, stats, bn, stream);

    dgcnn_pool_kernel<<<BATCH * 1024 / 256, 256, 0, stream>>>(bufA, 1024, pooledh);

    // FC1: 16x1024 @ 1024x512 (+ReLU, f16 out), FC2: 16x512 @ 512x40 -> d_out.
    dgcnn_wswizzle_kernel<<<(32 * 32 * 512 + 255) / 256, 256, 0, stream>>>(
        Wc1, 1024, 512, 32, Wsw, 32 * 32 * 512);
    dgcnn_fc_kernel<<<8, 128, 0, stream>>>(pooledh, Wsw, bc1, 1024, 512,
                                           nullptr, f1h);
    dgcnn_wswizzle_kernel<<<(3 * 16 * 512 + 255) / 256, 256, 0, stream>>>(
        Wc2, 512, 40, 16, Wsw, 3 * 16 * 512);
    dgcnn_fc_kernel<<<1, 128, 0, stream>>>(f1h, Wsw, bc2, 512, 40,
                                           (float*)d_out, nullptr);
}
